// EdgeGuidedFusionLoss_25529285607897
// MI455X (gfx1250) — matrix operand order repaired and checked by reading, not compiled, not run on hardware
//
#include <hip/hip_runtime.h>
#include <hip/hip_bf16.h>
#include <math.h>

typedef __attribute__((ext_vector_type(16))) _Float16 v16h;
typedef __attribute__((ext_vector_type(8)))  float    v8f;

#define BB 8
#define HH 512
#define WW 512
#define NPIX (HH*WW)
#define NPTS 4096
#define SEPS 0.0025f
#define SITERS 10
#define NPROB 24
#define SOUT 502
#define PH 256
#define PW 256

// raw v_exp_f32 (flush-to-zero below 2^-126 is fine for Gaussian kernel tails)
#if __has_builtin(__builtin_amdgcn_exp2f)
#define FAST_EXP2(x) __builtin_amdgcn_exp2f(x)
#else
#define FAST_EXP2(x) exp2f(x)
#endif

// ---- workspace layout (word offsets in header, byte offsets after) ----
#define ACC_L1M 0
#define ACC_L1P 1
#define ACC_TVY 2
#define ACC_TVX 3
#define ACC_LINT 4
#define ACC_SAL 5
#define ACC_G1 6
#define ACC_G2 7
#define ACC_AUX 8
#define ACC_SMOOTH 9
#define PMIN_OFF 16      // 8 uint
#define PMAX_OFF 24      // 8 uint
#define MAGMAX_OFF 32    // 4*8 uint: y, mri, ypool, mripool
#define CNT_OFF 64       // 8 float
#define SSIMPS_OFF 72    // 3*8 float
#define OUTSQ_OFF 96     // 8 float
#define SINKOT_OFF 104   // 24 float
#define MVEC_OFF 128     // 24 float
#define HDR_WORDS 256
#define F_OFF    (HDR_WORDS*4)
#define G_OFF    (F_OFF + NPROB*NPTS*4)
#define U_OFF    (G_OFF + NPROB*NPTS*4)
#define YPTS_OFF (U_OFF + NPROB*NPTS*2)
#define PPTS_OFF (YPTS_OFF + BB*NPTS*4)
#define PY_OFF   (PPTS_OFF + BB*NPTS*4)
#define PM_OFF   (PY_OFF + BB*PH*PW*4)

// ---------------- helpers ----------------
__device__ inline float blkSum(float v, float* sh) {
  sh[threadIdx.x] = v; __syncthreads();
  for (int s = 128; s > 0; s >>= 1) {
    if ((int)threadIdx.x < s) sh[threadIdx.x] += sh[threadIdx.x + s];
    __syncthreads();
  }
  float r = sh[0]; __syncthreads(); return r;
}
__device__ inline float blkMax(float v, float* sh) {
  sh[threadIdx.x] = v; __syncthreads();
  for (int s = 128; s > 0; s >>= 1) {
    if ((int)threadIdx.x < s) sh[threadIdx.x] = fmaxf(sh[threadIdx.x], sh[threadIdx.x + s]);
    __syncthreads();
  }
  float r = sh[0]; __syncthreads(); return r;
}
__device__ inline float blkMin(float v, float* sh) {
  sh[threadIdx.x] = v; __syncthreads();
  for (int s = 128; s > 0; s >>= 1) {
    if ((int)threadIdx.x < s) sh[threadIdx.x] = fminf(sh[threadIdx.x], sh[threadIdx.x + s]);
    __syncthreads();
  }
  float r = sh[0]; __syncthreads(); return r;
}
__device__ inline float sobelMag(const float* s, int Hs, int Ws, int i, int j) {
  float v[3][3];
  #pragma unroll
  for (int a = 0; a < 3; ++a)
    #pragma unroll
    for (int b = 0; b < 3; ++b) {
      int ii = i + a - 1, jj = j + b - 1;
      v[a][b] = (ii >= 0 && ii < Hs && jj >= 0 && jj < Ws) ? s[ii * Ws + jj] : 0.0f;
    }
  float gx = (v[0][0] - v[0][2]) + 2.0f * (v[1][0] - v[1][2]) + (v[2][0] - v[2][2]);
  float gy = (v[0][0] + 2.0f * v[0][1] + v[0][2]) - (v[2][0] + 2.0f * v[2][1] + v[2][2]);
  return sqrtf(gx * gx + gy * gy + 1e-6f);
}
__device__ inline int refl(int c, int n) { return c < 0 ? (-1 - c) : (c >= n ? (2 * n - 1 - c) : c); }

// gfx1250 async global->LDS 16B copy (ASYNCcnt-tracked); lptr must be a
// generic pointer into __shared__ (low 32 bits = LDS offset per aperture rule).
__device__ inline void asyncCopy16(const void* gptr, void* lptr) {
  unsigned loff = (unsigned)(uintptr_t)lptr;
  unsigned long long ga = (unsigned long long)(uintptr_t)gptr;
  asm volatile("global_load_async_to_lds_b128 %0, %1, off"
               :: "v"(loff), "v"(ga) : "memory");
}
__device__ inline void asyncWait() {
  asm volatile("s_wait_asynccnt 0x0" ::: "memory");
}

// ---------------- init ----------------
__global__ void k_init(unsigned* ws) {
  int total = HDR_WORDS + NPROB * NPTS * 2;   // header + f + g zeroed
  for (int i = blockIdx.x * blockDim.x + threadIdx.x; i < total; i += gridDim.x * blockDim.x) {
    unsigned v = 0u;
    if (i >= PMIN_OFF && i < PMIN_OFF + 8) v = 0x7F7FFFFFu;
    ws[i] = v;
  }
}

// ---------------- pooling ----------------
__global__ void k_pool(const float* y, const float* mri, char* ws) {
  int t = blockIdx.x * blockDim.x + threadIdx.x;
  if (t >= BB * PH * PW) return;
  int b = t / (PH * PW), r = t % (PH * PW);
  int i = r / PW, j = r % PW;
  int base = b * NPIX + (2 * i) * WW + 2 * j;
  float* py = (float*)(ws + PY_OFF);
  float* pm = (float*)(ws + PM_OFF);
  py[t] = 0.25f * (y[base] + y[base + 1] + y[base + WW] + y[base + WW + 1]);
  pm[t] = 0.25f * (mri[base] + mri[base + 1] + mri[base + WW] + mri[base + WW + 1]);
}

// ---------------- point gather ----------------
__global__ void k_gather(const float* y, const float* pet, const float* mask, const int* idx, char* ws) {
  int t = blockIdx.x * blockDim.x + threadIdx.x;
  if (t >= BB * NPTS) return;
  int b = t / NPTS;
  int p = b * NPIX + idx[t];
  ((float*)(ws + YPTS_OFF))[t] = y[p] * mask[p];
  ((float*)(ws + PPTS_OFF))[t] = pet[p] * mask[p];
}

// ---------------- fused elementwise reductions ----------------
__global__ __launch_bounds__(256) void k_elem(const float* y, const float* mri, const float* pet,
                                              const float* mask, char* ws) {
  __shared__ float sh[256];
  float* acc = (float*)ws;
  float* cnt = acc + CNT_OFF;
  float* outsq = acc + OUTSQ_OFF;
  unsigned* pmin = (unsigned*)ws + PMIN_OFF;
  unsigned* pmax = (unsigned*)ws + PMAX_OFF;
  float l1m = 0, l1p = 0, tvy = 0, tvx = 0, lint = 0;
  for (int p0 = blockIdx.x * 256; p0 < BB * NPIX; p0 += gridDim.x * 256) {
    int p = p0 + threadIdx.x;
    int b = p0 / NPIX;
    int r = (p / WW) % HH, c = p % WW;
    float yv = y[p], mv = mri[p], pv = pet[p], mk = mask[p];
    l1m += fabsf(yv - mv);
    l1p += fabsf(yv - pv);
    if (r < HH - 1) tvy += fabsf(y[p + WW] - yv);
    if (c < WW - 1) tvx += fabsf(y[p + 1] - yv);
    float dm = (yv - pv) * mk; lint += dm * dm;
    float o = yv * (1.0f - mk); o = o * o;
    float cv = (mk > 0.1f) ? 1.0f : 0.0f;
    float s;
    s = blkSum(o, sh);  if (threadIdx.x == 0) atomicAdd(&outsq[b], s);
    s = blkSum(cv, sh); if (threadIdx.x == 0) atomicAdd(&cnt[b], s);
    s = blkMax(pv, sh); if (threadIdx.x == 0) atomicMax(&pmax[b], __float_as_uint(s));
    s = blkMin(pv, sh); if (threadIdx.x == 0) atomicMin(&pmin[b], __float_as_uint(s));
  }
  float s;
  s = blkSum(l1m, sh); if (threadIdx.x == 0) atomicAdd(&acc[ACC_L1M], s);
  s = blkSum(l1p, sh); if (threadIdx.x == 0) atomicAdd(&acc[ACC_L1P], s);
  s = blkSum(tvy, sh); if (threadIdx.x == 0) atomicAdd(&acc[ACC_TVY], s);
  s = blkSum(tvx, sh); if (threadIdx.x == 0) atomicAdd(&acc[ACC_TVX], s);
  s = blkSum(lint, sh); if (threadIdx.x == 0) atomicAdd(&acc[ACC_LINT], s);
}

// ---------------- per-sample sobel magnitude max ----------------
__global__ __launch_bounds__(256) void k_sobelmax(const float* src, int Hs, int Ws, unsigned* mx) {
  __shared__ float sh[256];
  int tot = BB * Hs * Ws;
  for (int p0 = blockIdx.x * 256; p0 < tot; p0 += gridDim.x * 256) {
    int p = p0 + threadIdx.x;
    int b = p0 / (Hs * Ws);
    int r = (p / Ws) % Hs, c = p % Ws;
    float m = sobelMag(src + b * Hs * Ws, Hs, Ws, r, c);
    float s = blkMax(m, sh);
    if (threadIdx.x == 0) atomicMax(&mx[b], __float_as_uint(s));
  }
}

// ---------------- grad consistency (+ aux) ----------------
__global__ __launch_bounds__(256) void k_grad(const float* sa, const float* sb,
                                              const unsigned* mxa, const unsigned* mxb,
                                              int Hs, int Ws, char* ws, int accIdx,
                                              const float* eh, int auxIdx) {
  __shared__ float sh[256];
  float* acc = (float*)ws;
  float g = 0, ax = 0;
  int tot = BB * Hs * Ws;
  for (int p0 = blockIdx.x * 256; p0 < tot; p0 += gridDim.x * 256) {
    int p = p0 + threadIdx.x;
    int b = p0 / (Hs * Ws);
    int r = (p / Ws) % Hs, c = p % Ws;
    float ia = 1.0f / fmaxf(__uint_as_float(mxa[b]), 1e-6f);
    float ib = 1.0f / fmaxf(__uint_as_float(mxb[b]), 1e-6f);
    float ma = fminf(fmaxf(sobelMag(sa + b * Hs * Ws, Hs, Ws, r, c) * ia, 0.0f), 1.0f);
    float mb = fminf(fmaxf(sobelMag(sb + b * Hs * Ws, Hs, Ws, r, c) * ib, 0.0f), 1.0f);
    float d = ma - mb;
    g += sqrtf(d * d + 1e-6f);
    if (eh) ax += fabsf(eh[p] - mb);
  }
  float s;
  s = blkSum(g, sh); if (threadIdx.x == 0) atomicAdd(&acc[accIdx], s);
  if (eh) { s = blkSum(ax, sh); if (threadIdx.x == 0) atomicAdd(&acc[auxIdx], s); }
}

// ---------------- PET saliency L1 ----------------
__global__ __launch_bounds__(256) void k_sal(const float* y, const float* pet, char* ws) {
  __shared__ float sh[256];
  float* acc = (float*)ws;
  unsigned* pmin = (unsigned*)ws + PMIN_OFF;
  unsigned* pmax = (unsigned*)ws + PMAX_OFF;
  float s = 0;
  for (int p0 = blockIdx.x * 256; p0 < BB * NPIX; p0 += gridDim.x * 256) {
    int p = p0 + threadIdx.x;
    int b = p0 / NPIX;
    float pv = pet[p];
    float pmn = __uint_as_float(pmin[b]), pmx = __uint_as_float(pmax[b]);
    float pn = (pv - pmn) / (pmx - pmn + 1e-6f);
    float S = 1.0f / (1.0f + expf(-6.0f * (pn - 0.4f)));
    s += S * fabsf(y[p] - pv);
  }
  float r = blkSum(s, sh);
  if (threadIdx.x == 0) atomicAdd(&acc[ACC_SAL], r);
}

// ---------------- gaussian smooth L2 ----------------
__global__ __launch_bounds__(256) void k_smooth(const float* y, char* ws) {
  __shared__ float sh[256];
  float* acc = (float*)ws;
  float w[9];
  {
    float sm = 0;
    #pragma unroll
    for (int k = 0; k < 9; ++k) { float d = (float)(k - 4); w[k] = expf(-0.5f * d * d); sm += w[k]; }
    #pragma unroll
    for (int k = 0; k < 9; ++k) w[k] /= sm;
  }
  float s = 0;
  for (int p0 = blockIdx.x * 256; p0 < BB * NPIX; p0 += gridDim.x * 256) {
    int p = p0 + threadIdx.x;
    int b = p0 / NPIX;
    int r = (p / WW) % HH, c = p % WW;
    const float* yb = y + b * NPIX;
    float a = 0;
    #pragma unroll
    for (int u = 0; u < 9; ++u) {
      int ii = refl(r + u - 4, HH);
      float rowacc = 0;
      #pragma unroll
      for (int v = 0; v < 9; ++v) rowacc += w[v] * yb[ii * WW + refl(c + v - 4, WW)];
      a += w[u] * rowacc;
    }
    float d = a - y[p];
    s += d * d;
  }
  float r = blkSum(s, sh);
  if (threadIdx.x == 0) atomicAdd(&acc[ACC_SMOOTH], r);
}

// ---------------- SSIM (11x11 gaussian window, VALID) ----------------
__global__ __launch_bounds__(256) void k_ssim(const float* x, const float* yr, const float* msk,
                                              char* ws, int psIdx) {
  __shared__ float xs[26][28], ys2[26][28], wv[11];
  __shared__ float sh[256];
  int b = blockIdx.z;
  if (threadIdx.x == 0) {
    float sm = 0;
    for (int k = 0; k < 11; ++k) { float d = (float)(k - 5); wv[k] = expf(-d * d / 4.5f); sm += wv[k]; }
    for (int k = 0; k < 11; ++k) wv[k] /= sm;
  }
  __syncthreads();
  int oy0 = blockIdx.y * 16, ox0 = blockIdx.x * 16;
  const float* xb = x + b * NPIX;
  const float* yb = yr + b * NPIX;
  const float* mb = msk ? msk + b * NPIX : nullptr;
  for (int idx = threadIdx.x; idx < 26 * 26; idx += 256) {
    int r = idx / 26, c = idx % 26;
    int gi = min(oy0 + r, HH - 1), gj = min(ox0 + c, WW - 1);
    float m = mb ? mb[gi * WW + gj] : 1.0f;
    xs[r][c]  = xb[gi * WW + gj] * m;
    ys2[r][c] = yb[gi * WW + gj] * m;
  }
  __syncthreads();
  int ty = threadIdx.x / 16, tx = threadIdx.x % 16;
  int oy = oy0 + ty, ox = ox0 + tx;
  float val = 0;
  if (oy < SOUT && ox < SOUT) {
    float sx = 0, sy = 0, sxx = 0, syy = 0, sxy = 0;
    #pragma unroll 1
    for (int a = 0; a < 11; ++a) {
      float wa = wv[a];
      #pragma unroll
      for (int bq = 0; bq < 11; ++bq) {
        float w2 = wa * wv[bq];
        float xv = xs[ty + a][tx + bq], yv = ys2[ty + a][tx + bq];
        sx += w2 * xv; sy += w2 * yv;
        sxx += w2 * xv * xv; syy += w2 * yv * yv; sxy += w2 * xv * yv;
      }
    }
    float vx = sxx - sx * sx, vy = syy - sy * sy, cxy = sxy - sx * sy;
    const float C1 = 1e-4f, C2 = 9e-4f;
    val = ((2.0f * sx * sy + C1) * (2.0f * cxy + C2)) /
          ((sx * sx + sy * sy + C1) * (vx + vy + C2));
  }
  float r = blkSum(val, sh);
  if (threadIdx.x == 0) atomicAdd(&((float*)ws)[SSIMPS_OFF + psIdx * 8 + b], r);
}

// ---------------- Sinkhorn: stabilization prep (max + u = exp((v-m)/eps)) ----------------
__global__ __launch_bounds__(256) void k_sink_prep(char* ws, int phase) {
  int t = blockIdx.x;
  const float* v = (const float*)(ws + (phase == 0 ? F_OFF : G_OFF)) + t * NPTS;
  _Float16* u = (_Float16*)(ws + U_OFF) + t * NPTS;
  __shared__ float sh[256];
  float m = -3.4e38f;
  for (int i = threadIdx.x; i < NPTS; i += 256) m = fmaxf(m, v[i]);
  m = blkMax(m, sh);
  if (threadIdx.x == 0) ((float*)ws)[MVEC_OFF + t] = m;
  const float CE = (1.0f / SEPS) * 1.4426950408889634f;
  for (int i = threadIdx.x; i < NPTS; i += 256) u[i] = (_Float16)FAST_EXP2((v[i] - m) * CE);
}

// ---------------- Sinkhorn: WMMA kernel-matvec half-iteration ----------------
// out_p = -m_s + eps*ln(n) - eps*ln( sum_q u_q * exp(-0.5*(Q_q - P_p)^2/eps) )
__global__ __launch_bounds__(256) void k_sink_update(char* ws, int phase) {
  int t = blockIdx.y;
  int kind = t % 3, b = t / 3;
  const float* ypts = (const float*)(ws + YPTS_OFF) + b * NPTS;
  const float* ppts = (const float*)(ws + PPTS_OFF) + b * NPTS;
  const float* xa = (kind == 2) ? ppts : ypts;
  const float* xb = (kind == 1) ? ypts : ppts;
  const float* Q = (phase == 0) ? xa : xb;   // reduction axis points
  const float* P = (phase == 0) ? xb : xa;   // output axis points
  float* out = (float*)(ws + (phase == 0 ? G_OFF : F_OFF)) + t * NPTS;
  const _Float16* u = (const _Float16*)(ws + U_OFF) + t * NPTS;
  float ms = ((const float*)ws)[MVEC_OFF + t];

  __shared__ alignas(16) float Qs[NPTS];
  __shared__ alignas(16) _Float16 us[NPTS];
  // async global->LDS staging: Q = 1024 x 16B, u = 512 x 16B
  for (int i = threadIdx.x; i < 1024; i += 256)
    asyncCopy16((const char*)Q + i * 16, (char*)Qs + i * 16);
  for (int i = threadIdx.x; i < 512; i += 256)
    asyncCopy16((const char*)u + i * 16, (char*)us + i * 16);
  asyncWait();
  __syncthreads();

  int lane = threadIdx.x & 31;
  int wave = threadIdx.x >> 5;
  int p0 = blockIdx.x * 128 + wave * 16;
  int n = lane & 15;
  int h = lane >> 4;
  float Pp = P[p0 + n];
  const float CE = (-0.5f / SEPS) * 1.4426950408889634f;  // exp(-0.5 d^2/eps) via exp2

  const unsigned* us32 = (const unsigned*)us;
  union UA { v16h v; unsigned u32[8]; int4 q[2]; };

  v8f acc = {};
  #pragma unroll 2
  for (int q0 = 0; q0 < NPTS; q0 += 32) {
    // A fragment (16x32 f16): every row = same u chunk (rows 1..15 of D ignored).
    // Lane layout: element e <-> K = 8h + e + (e&8): two contiguous 16B runs.
    UA A;
    int bse = (q0 >> 1) + 4 * h;               // dword index of u[q0 + 8h]
    A.q[0] = *(const int4*)(us32 + bse);       // K = 8h+0..7
    A.q[1] = *(const int4*)(us32 + bse + 8);   // K = 8h+16..23
    // B fragment (32x16 f16): lane col n, element e <-> K = 8h + e + (e&8) (same as A).
    int qb0 = q0 + 8 * h;
    float4 qv0 = *(const float4*)(Qs + qb0);
    float4 qv1 = *(const float4*)(Qs + qb0 + 4);
    float4 qv2 = *(const float4*)(Qs + qb0 + 16);
    float4 qv3 = *(const float4*)(Qs + qb0 + 20);
    float qv[16] = { qv0.x, qv0.y, qv0.z, qv0.w, qv1.x, qv1.y, qv1.z, qv1.w,
                     qv2.x, qv2.y, qv2.z, qv2.w, qv3.x, qv3.y, qv3.z, qv3.w };
    v16h bf;
    #pragma unroll
    for (int e = 0; e < 16; ++e) {
      float d = qv[e] - Pp;
      bf[e] = (_Float16)FAST_EXP2(d * d * CE);
    }
    acc = __builtin_amdgcn_wmma_f32_16x16x32_f16(false, A.v, false, bf, (short)0, acc,
                                                 false, false);
  }
  if (lane < 16) {                             // D row M=0: VGPR0, lanes 0..15 = N
    float S = fmaxf(acc[0], 1e-30f);
    out[p0 + lane] = -ms + SEPS * logf((float)NPTS) - SEPS * logf(S);
  }
}

// ---------------- Sinkhorn final: mean(f)+mean(g) ----------------
__global__ __launch_bounds__(256) void k_sinkfin(char* ws) {
  int t = blockIdx.x;
  const float* f = (const float*)(ws + F_OFF) + t * NPTS;
  const float* g = (const float*)(ws + G_OFF) + t * NPTS;
  __shared__ float sh[256];
  float s = 0;
  for (int i = threadIdx.x; i < NPTS; i += 256) s += f[i] + g[i];
  float r = blkSum(s, sh);
  if (threadIdx.x == 0) ((float*)ws)[SINKOT_OFF + t] = r / (float)NPTS;
}

// ---------------- finalize ----------------
__global__ void k_final(char* ws, float* out) {
  if (threadIdx.x != 0 || blockIdx.x != 0) return;
  const float* acc = (const float*)ws;
  const float NP = (float)(BB * NPIX);
  const float NSS = (float)(SOUT * SOUT);
  float ssim_m_sum = 0, ssim_p_sum = 0;
  float Lt = 0, Lo = 0, Ls = 0;
  for (int b = 0; b < BB; ++b) {
    ssim_m_sum += acc[SSIMPS_OFF + 0 * 8 + b];
    ssim_p_sum += acc[SSIMPS_OFF + 1 * 8 + b];
    float wv = (acc[CNT_OFF + b] >= 32.0f) ? 1.0f : 0.0f;
    float ot = acc[SINKOT_OFF + 3 * b + 0]
             - 0.5f * (acc[SINKOT_OFF + 3 * b + 1] + acc[SINKOT_OFF + 3 * b + 2]);
    Lt += wv * ot;
    Lo += wv * (acc[OUTSQ_OFF + b] / (float)NPIX);
    Ls += wv * (1.0f - acc[SSIMPS_OFF + 2 * 8 + b] / NSS);
  }
  Lt /= BB; Lo /= BB; Ls /= BB;
  float ssim_m = 1.0f - ssim_m_sum / (BB * NSS);
  float ssim_p = 1.0f - ssim_p_sum / (BB * NSS);
  float l1_m = acc[ACC_L1M] / NP;
  float l1_p = acc[ACC_L1P] / NP;
  float sal = acc[ACC_SAL] / NP;
  float tv = acc[ACC_TVY] / (float)(BB * (HH - 1) * WW) + acc[ACC_TVX] / (float)(BB * HH * (WW - 1));
  float g_cons = acc[ACC_G1] / NP + acc[ACC_G2] / (float)(BB * PH * PW);
  float aux = acc[ACC_AUX] / NP;
  float Lint = 2.5f * acc[ACC_LINT] / NP;
  float Lsm = acc[ACC_SMOOTH] / NP;
  float loss = 1.0f * ssim_m + 0.8f * Lt + 0.3f * Lo + 0.5f * Ls + Lint;
  out[0] = loss;  out[1] = ssim_m; out[2] = ssim_p; out[3] = g_cons; out[4] = aux;
  out[5] = Lsm;   out[6] = tv;     out[7] = l1_m;   out[8] = l1_p;   out[9] = sal;
}

// ---------------- host launcher ----------------
extern "C" void kernel_launch(void* const* d_in, const int* in_sizes, int n_in,
                              void* d_out, int out_size, void* d_ws, size_t ws_size,
                              hipStream_t stream) {
  (void)in_sizes; (void)n_in; (void)out_size; (void)ws_size;
  const float* y    = (const float*)d_in[0];
  const float* mri  = (const float*)d_in[1];
  const float* pet  = (const float*)d_in[2];
  const float* mask = (const float*)d_in[3];
  const float* ehat = (const float*)d_in[4];
  const int*   idx  = (const int*)d_in[5];
  char* ws = (char*)d_ws;
  float* out = (float*)d_out;
  unsigned* mag = (unsigned*)ws + MAGMAX_OFF;

  k_init<<<256, 256, 0, stream>>>((unsigned*)ws);
  k_pool<<<(BB * PH * PW + 255) / 256, 256, 0, stream>>>(y, mri, ws);
  k_gather<<<(BB * NPTS + 255) / 256, 256, 0, stream>>>(y, pet, mask, idx, ws);
  k_elem<<<2048, 256, 0, stream>>>(y, mri, pet, mask, ws);
  k_sobelmax<<<1024, 256, 0, stream>>>(y, HH, WW, mag + 0);
  k_sobelmax<<<1024, 256, 0, stream>>>(mri, HH, WW, mag + 8);
  k_sobelmax<<<512, 256, 0, stream>>>((const float*)(ws + PY_OFF), PH, PW, mag + 16);
  k_sobelmax<<<512, 256, 0, stream>>>((const float*)(ws + PM_OFF), PH, PW, mag + 24);
  k_grad<<<1024, 256, 0, stream>>>(y, mri, mag + 0, mag + 8, HH, WW, ws, ACC_G1, ehat, ACC_AUX);
  k_grad<<<512, 256, 0, stream>>>((const float*)(ws + PY_OFF), (const float*)(ws + PM_OFF),
                                  mag + 16, mag + 24, PH, PW, ws, ACC_G2, nullptr, 0);
  k_sal<<<1024, 256, 0, stream>>>(y, pet, ws);
  k_smooth<<<1024, 256, 0, stream>>>(y, ws);
  {
    dim3 g(32, 32, BB);
    k_ssim<<<g, 256, 0, stream>>>(y, mri, nullptr, ws, 0);
    k_ssim<<<g, 256, 0, stream>>>(y, pet, nullptr, ws, 1);
    k_ssim<<<g, 256, 0, stream>>>(y, mri, mask, ws, 2);
  }
  // Sinkhorn: 10 iterations of (g-update from f) then (f-update from g)
  dim3 ug(NPTS / 128, NPROB);
  for (int it = 0; it < SITERS; ++it) {
    k_sink_prep<<<NPROB, 256, 0, stream>>>(ws, 0);
    k_sink_update<<<ug, 256, 0, stream>>>(ws, 0);
    k_sink_prep<<<NPROB, 256, 0, stream>>>(ws, 1);
    k_sink_update<<<ug, 256, 0, stream>>>(ws, 1);
  }
  k_sinkfin<<<NPROB, 256, 0, stream>>>(ws);
  k_final<<<1, 32, 0, stream>>>(ws, out);
}
